// CTNN_15023795601630
// MI455X (gfx1250) — matrix operand order repaired and checked by reading, not compiled, not run on hardware
//
#include <hip/hip_runtime.h>
#include <hip/hip_bf16.h>

typedef __attribute__((ext_vector_type(16))) _Float16 v16h;
typedef __attribute__((ext_vector_type(8)))  _Float16 v8h;
typedef __attribute__((ext_vector_type(8)))  float    v8f;

// ---------------- problem constants ----------------
constexpr int T = 15;
constexpr int C1 = 30;          // layer-1 feature maps
constexpr int C2 = 100;         // layer-2 feature maps
constexpr int HIN = 320, WIN = 320;
constexpr int HP = 162, WP = 162;      // pooled+padded spike input to conv2
constexpr int HO = 158, WO = 158;      // conv2 output
constexpr int K2 = 750;                // 30*5*5
constexpr int KSTEPS = 24;             // ceil(750/32)
constexpr int MTILES = 7;              // ceil(100/16)
constexpr size_t HW2 = (size_t)HO * WO;            // 24964
constexpr size_t NEL = (size_t)C2 * HW2;           // 2,496,400
constexpr size_t S2  = (size_t)T * NEL;            // 37,446,000 (spk2 / pot2 each)

// ---------------- workspace layout (bytes) ----------------
constexpr size_t align256(size_t x) { return (x + 255) & ~(size_t)255; }
constexpr size_t SPKIN_BYTES = (size_t)T * C1 * HP * WP * 2;           // f16
constexpr size_t SPKIN_OFF   = 0;
constexpr size_t APRE_OFF    = align256(SPKIN_OFF + SPKIN_BYTES);
constexpr size_t APRE_BYTES  = (size_t)MTILES * KSTEPS * 32 * 16 * 2;  // f16 fragments
constexpr size_t VALUE_OFF   = align256(APRE_OFF + APRE_BYTES);
constexpr size_t VALUE_BYTES = NEL * 4;
constexpr size_t NSPK_OFF    = align256(VALUE_OFF + VALUE_BYTES);
constexpr size_t NSPK_BYTES  = NEL * 4;
constexpr size_t VMAX_OFF    = align256(NSPK_OFF + NSPK_BYTES);

// ============================================================
// K0: swizzle w2 [100][750] f32 -> WMMA A-fragments (f16)
// A-matrix 16x32 f16 layout (ISA 7.12.2): lane = (M row)&15 | half<<4
//   VGPR v<4 : K = 2v+{0,1} + 8*half ;  v>=4 : K = 16+2(v-4)+{0,1} + 8*half
// ============================================================
__global__ void ctnn_prep_w2(const float* __restrict__ w2, _Float16* __restrict__ apre) {
  int mt   = blockIdx.x;            // 0..6
  int ks   = blockIdx.y;            // 0..23
  int lane = threadIdx.x;           // 0..31
  int row  = mt * 16 + (lane & 15);
  int half = lane >> 4;
  _Float16* dst = apre + (((size_t)mt * KSTEPS + ks) * 32 + lane) * 16;
#pragma unroll
  for (int j = 0; j < 16; ++j) {
    int v = j >> 1;
    int klocal = ((v < 4) ? (2 * v) : (16 + 2 * (v - 4))) + (j & 1) + half * 8;
    int k = ks * 32 + klocal;
    float x = 0.0f;
    if (row < C2 && k < K2) x = w2[(size_t)row * K2 + k];
    dst[j] = (_Float16)x;
  }
}

// ============================================================
// K1: conv1 (pad 2) + threshold@15 + 2x2 pooled OR -> padded f16 spikes
// ============================================================
__global__ void ctnn_layer1_pool(const float* __restrict__ in,
                                 const float* __restrict__ w1,
                                 _Float16* __restrict__ spkin) {
  int t = blockIdx.z, c = blockIdx.y;
  __shared__ float wc[50];
  if (threadIdx.x < 50) wc[threadIdx.x] = w1[c * 50 + threadIdx.x];
  __syncthreads();
  int pix = blockIdx.x * blockDim.x + threadIdx.x;   // 0..25599
  int ph = pix / 160, pw = pix % 160;
  const float* ib = in + (size_t)t * 2 * HIN * WIN;
  float spike = 0.0f;
#pragma unroll
  for (int dy = 0; dy < 2; ++dy)
#pragma unroll
    for (int dx = 0; dx < 2; ++dx) {
      int y = 2 * ph + dy, x = 2 * pw + dx;          // conv1 output coords
      float pot = 0.0f;
#pragma unroll
      for (int ci = 0; ci < 2; ++ci)
#pragma unroll
        for (int kh = 0; kh < 5; ++kh) {
          int sy = y + kh - 2;
          if (sy < 0 || sy >= HIN) continue;
#pragma unroll
          for (int kw = 0; kw < 5; ++kw) {
            int sx = x + kw - 2;
            if (sx < 0 || sx >= WIN) continue;
            pot += ib[((size_t)ci * HIN + sy) * WIN + sx] * wc[ci * 25 + kh * 5 + kw];
          }
        }
      if (pot > 15.0f) spike = 1.0f;
    }
  spkin[(((size_t)t * C1 + c) * HP + (ph + 1)) * WP + (pw + 1)] = (_Float16)spike;
}

// ============================================================
// K2: conv2 as f16 WMMA GEMM.
//  - async global->LDS copy of the 150 raw patch rows (gfx1250 TDM-class path)
//  - branch-free LDS im2col into column-major xcol[n][k] (k-tail pre-zeroed)
//  - GEMM loop: contiguous 16B LDS loads for B, coalesced 32B/lane global A
// ============================================================
__global__ __launch_bounds__(224)
void ctnn_layer2_wmma(const _Float16* __restrict__ spkin,
                      const _Float16* __restrict__ apre,
                      float* __restrict__ spk2, float* __restrict__ pot2) {
  constexpr int XSTR = 776;                 // 768 + 8 pad: bank step 4 -> conflict-free
  __shared__ _Float16 rows_sh[150 * 20];    // [ci*5+kh][20] raw patch rows
  __shared__ _Float16 xcol[16 * XSTR];      // column-major im2col [n][k]
  const int t = blockIdx.z;
  const int oh = blockIdx.y;
  const int ow0 = blockIdx.x * 16;
  const int tid = threadIdx.x;

  // ---- stage raw rows: 1500 dwords via async global->LDS ----
  const _Float16* sp = spkin + (size_t)t * C1 * HP * WP;
  const unsigned ldsbase = (unsigned)(uintptr_t)(_Float16*)rows_sh;
  for (int d = tid; d < 1500; d += 224) {
    int rr = d / 10;                        // ci*5+kh
    int dw = d % 10;                        // dword within 20-half row
    int ci = rr / 5, kh = rr % 5;
    const _Float16* g = sp + ((size_t)ci * HP + (oh + kh)) * WP + ow0 + dw * 2;
    unsigned lds = ldsbase + (unsigned)(rr * 40 + dw * 4);
    unsigned long long ga = (unsigned long long)(uintptr_t)g;
    asm volatile("global_load_async_to_lds_b32 %0, %1, off"
                 :: "v"(lds), "v"(ga) : "memory");
  }
  asm volatile("s_wait_asynccnt 0x0" ::: "memory");
  __syncthreads();

  // ---- branch-free im2col: rows -> xcol[n][k], zero k in [750,768) ----
  for (int i = tid; i < 768 * 16; i += 224) {
    int k = i >> 4;
    int n = i & 15;
    _Float16 v = (_Float16)0.0f;
    if (k < K2) {
      int ci = k / 25, r25 = k % 25;
      v = rows_sh[(ci * 5 + r25 / 5) * 20 + (n + r25 % 5)];
    }
    xcol[n * XSTR + k] = v;
  }
  __syncthreads();

  const int lane = tid & 31;
  const int wid  = tid >> 5;                // m-tile 0..6
  const int half = lane >> 4;
  const int n    = lane & 15;

  v8f acc = {};
  const _Float16* ap = apre + (((size_t)wid * KSTEPS) * 32 + lane) * 16;
  const _Float16* xp = &xcol[n * XSTR + half * 8];

  for (int ks = 0; ks < KSTEPS; ++ks) {
    v16h a = *(const v16h*)(ap + (size_t)ks * 512);
    if (ks + 1 < KSTEPS)
      __builtin_prefetch(ap + (size_t)(ks + 1) * 512, 0, 0);
    v8h blo = *(const v8h*)(xp + ks * 32);
    v8h bhi = *(const v8h*)(xp + ks * 32 + 16);
    v16h b = __builtin_shufflevector(blo, bhi,
                                     0, 1, 2, 3, 4, 5, 6, 7,
                                     8, 9, 10, 11, 12, 13, 14, 15);
    acc = __builtin_amdgcn_wmma_f32_16x16x32_f16(
        /*neg_a=*/false, a, /*neg_b=*/false, b,
        /*c_mod=*/(short)0, acc, /*reuse_a=*/false, /*reuse_b=*/false);
  }

  const int ow = ow0 + n;
#pragma unroll
  for (int r = 0; r < 8; ++r) {
    int m = wid * 16 + r + half * 8;
    if (m < C2 && ow < WO) {
      float p = acc[r];
      float thr = (p > 10.0f) ? p : 0.0f;
      size_t idx = (((size_t)t * C2 + m) * HO + oh) * WO + ow;
      spk2[idx] = (thr > 0.0f) ? 1.0f : 0.0f;
      pot2[idx] = thr;
    }
  }
}

// ============================================================
// K3: pointwise inhibition on pot2 (in place), one thread per pixel
// ============================================================
__global__ void ctnn_inhibit(float* __restrict__ pot) {
  int pix = blockIdx.x * blockDim.x + threadIdx.x;
  if (pix >= (int)HW2) return;
  float* base = pot + pix;
  int amax[T];
  int clampsum = 0, lastclamp = 0;
  for (int t = 0; t < T; ++t) {
    float mx = base[(size_t)t * NEL];
    int mi = 0;
    for (int f = 1; f < C2; ++f) {
      float v = base[(size_t)t * NEL + (size_t)f * HW2];
      if (v > mx) { mx = v; mi = f; }
    }
    amax[t] = mi;
    int cl = (mx > 0.0f) ? 1 : 0;
    clampsum += cl;
    if (t == T - 1) lastclamp = cl;
  }
  int earliest = T - clampsum;
  if (earliest < 0) earliest = 0;
  if (earliest > T - 1) earliest = T - 1;
  int winner = 0;
#pragma unroll
  for (int t = 0; t < T; ++t) winner = (t == earliest) ? amax[t] : winner;
  for (int t = 0; t < T; ++t)
    for (int f = 0; f < C2; ++f) {
      size_t idx = (size_t)t * NEL + (size_t)f * HW2;
      float v = base[idx];
      base[idx] = (lastclamp && f == winner) ? v : 0.0f;
    }
}

// ============================================================
// K4a: per-(f,h,w): spike count + earliest value; global max via uint atomicMax
// ============================================================
__global__ void ctnn_kwin_prep(const float* __restrict__ pot,
                               float* __restrict__ value, float* __restrict__ nspk,
                               unsigned int* __restrict__ vmaxbits) {
  size_t i = (size_t)blockIdx.x * blockDim.x + threadIdx.x;
  if (i >= NEL) return;
  float vals[T];
  int ns = 0;
#pragma unroll
  for (int t = 0; t < T; ++t) {
    float p = pot[(size_t)t * NEL + i];
    vals[t] = p;
    ns += (p > 0.0f) ? 1 : 0;
  }
  int e = T - ns;
  if (e < 0) e = 0;
  if (e > T - 1) e = T - 1;
  float val = 0.0f;
#pragma unroll
  for (int t = 0; t < T; ++t) val = (t == e) ? vals[t] : val;
  value[i] = val;
  nspk[i] = (float)ns;
  float cand = (ns > 0) ? val : 0.0f;       // values >= 0 -> uint order == float order
  atomicMax(vmaxbits, __float_as_uint(cand));
}

// K4b: total = nspk * (value + T*vmax), overwrites `value`
__global__ void ctnn_kwin_total(float* __restrict__ value,
                                const float* __restrict__ nspk,
                                const unsigned int* __restrict__ vmaxbits) {
  size_t i = (size_t)blockIdx.x * blockDim.x + threadIdx.x;
  if (i >= NEL) return;
  float vmax = __uint_as_float(*vmaxbits);
  value[i] = nspk[i] * (value[i] + vmax * (float)T);
}

// ============================================================
// K5: 8-winner iterative argmax with feature + columnar suppression
// ============================================================
__global__ __launch_bounds__(1024)
void ctnn_kwinners(float* __restrict__ total, float* __restrict__ out) {
  __shared__ float smax[1024];
  __shared__ int   sidx[1024];
  const int tid = threadIdx.x;
  for (int iter = 0; iter < 8; ++iter) {
    float mx = -1.0f; int mi = 0;
    for (size_t i = tid; i < NEL; i += 1024) {
      float v = total[i];
      if (v > mx) { mx = v; mi = (int)i; }
    }
    smax[tid] = mx; sidx[tid] = mi;
    __syncthreads();
    for (int s = 512; s > 0; s >>= 1) {
      if (tid < s) {
        bool take = (smax[tid + s] > smax[tid]) ||
                    (smax[tid + s] == smax[tid] && sidx[tid + s] < sidx[tid]);
        if (take) { smax[tid] = smax[tid + s]; sidx[tid] = sidx[tid + s]; }
      }
      __syncthreads();
    }
    float mval = smax[0];
    int idx = sidx[0];
    int f = idx / (int)HW2;
    int rem = idx % (int)HW2;
    int r = rem / WO, c = rem % WO;
    if (tid == 0) {
      if (mval != 0.0f) {
        out[iter * 3 + 0] = (float)f;
        out[iter * 3 + 1] = (float)r;
        out[iter * 3 + 2] = (float)c;
      } else {
        out[iter * 3 + 0] = -1.0f;
        out[iter * 3 + 1] = -1.0f;
        out[iter * 3 + 2] = -1.0f;
      }
    }
    __syncthreads();
    if (mval != 0.0f) {
      for (size_t i = tid; i < HW2; i += 1024) total[(size_t)f * HW2 + i] = 0.0f;
      for (int i = tid; i < C2 * 25; i += 1024) {
        int ff = i / 25;
        int rr = (i % 25) / 5 + r - 2;
        int cc = (i % 5) + c - 2;
        if (rr >= 0 && rr < HO && cc >= 0 && cc < WO)
          total[(size_t)ff * HW2 + (size_t)rr * WO + cc] = 0.0f;
      }
    }
    __syncthreads();
  }
}

// ============================================================
extern "C" void kernel_launch(void* const* d_in, const int* in_sizes, int n_in,
                              void* d_out, int out_size, void* d_ws, size_t ws_size,
                              hipStream_t stream) {
  const float* input = (const float*)d_in[0];   // [15,2,320,320]
  const float* w1    = (const float*)d_in[1];   // [30,2,5,5]
  const float* w2    = (const float*)d_in[2];   // [100,30,5,5]
  // d_in[3] = max_layer (assumed 2 per setup_inputs)

  float* out = (float*)d_out;
  char*  ws  = (char*)d_ws;
  _Float16* spkin = (_Float16*)(ws + SPKIN_OFF);
  _Float16* apre  = (_Float16*)(ws + APRE_OFF);
  float* value = (float*)(ws + VALUE_OFF);
  float* nspk  = (float*)(ws + NSPK_OFF);
  unsigned int* vmax = (unsigned int*)(ws + VMAX_OFF);

  float* spk2    = out;
  float* pot2    = out + S2;
  float* winners = out + 2 * S2;

  hipMemsetAsync(spkin, 0, SPKIN_BYTES, stream);
  hipMemsetAsync(vmax, 0, sizeof(unsigned int), stream);

  ctnn_prep_w2<<<dim3(MTILES, KSTEPS, 1), 32, 0, stream>>>(w2, apre);
  ctnn_layer1_pool<<<dim3(100, C1, T), 256, 0, stream>>>(input, w1, spkin);
  ctnn_layer2_wmma<<<dim3(10, HO, T), 224, 0, stream>>>(spkin, apre, spk2, pot2);
  ctnn_inhibit<<<dim3((unsigned)((HW2 + 255) / 256)), 256, 0, stream>>>(pot2);
  ctnn_kwin_prep<<<dim3((unsigned)((NEL + 255) / 256)), 256, 0, stream>>>(pot2, value, nspk, vmax);
  ctnn_kwin_total<<<dim3((unsigned)((NEL + 255) / 256)), 256, 0, stream>>>(value, nspk, vmax);
  ctnn_kwinners<<<1, 1024, 0, stream>>>(value, winners);
}
